// MatrixMergerBlock_28449863368926
// MI455X (gfx1250) — compile-verified, matrix operand order checked by reading, not run on hardware
//
#include <hip/hip_runtime.h>
#include <hip/hip_bf16.h>

// ---------------------------------------------------------------------------
// MatrixMergerBlock for MI455X (gfx1250, wave32, WMMA)
//   B=64, HP=64, H=64, W=64, D=32, NH=2, DH=16, DFF=64, R=1
// ---------------------------------------------------------------------------

#define B_   64
#define HP_  64
#define H_   64
#define W_   64
#define D_   32
#define DFF_ 64
#define NTOK (B_ * HP_)        // 4096 (b,hp) pairs
#define NPIX (H_ * W_)         // 4096 elements per pair

typedef __attribute__((ext_vector_type(16))) _Float16 v16h;
typedef __attribute__((ext_vector_type(8)))  float    v8f;

// ---------------------------------------------------------------------------
// WMMA fragment helpers (16x16x32 f16 -> f32)
//
// 16-bit A matrix 16x32 layout (per CDNA5 ISA 7.12.2):
//   lane L: M = L%16, half = L/16
//   packed pair p (0..7): Kbase = (p&3)*2 + (p>>2)*16 + half*8
// B matrix (32x16) is symmetric with N = L%16 (we compute X @ W^T, W row-major
// [N][K], so B[k][n] = W[n][k]).
// C/D f32 16x16: lane L holds N = L%16, rows M = r + 8*(L/16) for r=0..7.
// ---------------------------------------------------------------------------

__device__ __forceinline__ v16h frag_a_lds(const _Float16* S, int ld, int row0,
                                           int koff, int lane) {
  v16h f;
  int r = row0 + (lane & 15);
  int half = lane >> 4;
#pragma unroll
  for (int p = 0; p < 8; ++p) {
    int k = koff + ((p & 3) << 1) + ((p >> 2) << 4) + (half << 3);
    f[2 * p]     = S[r * ld + k];
    f[2 * p + 1] = S[r * ld + k + 1];
  }
  return f;
}

__device__ __forceinline__ v16h frag_b_wgt(const float* W, int ld, int n0,
                                           int koff, int lane) {
  v16h f;
  int n = n0 + (lane & 15);
  int half = lane >> 4;
#pragma unroll
  for (int p = 0; p < 8; ++p) {
    int k = koff + ((p & 3) << 1) + ((p >> 2) << 4) + (half << 3);
    f[2 * p]     = (_Float16)W[n * ld + k];
    f[2 * p + 1] = (_Float16)W[n * ld + k + 1];
  }
  return f;
}

__device__ __forceinline__ v8f wmma16(v16h a, v16h b, v8f c) {
  return __builtin_amdgcn_wmma_f32_16x16x32_f16(false, a, false, b, (short)0, c,
                                                false, false);
}

// ---------------------------------------------------------------------------
// Kernel 1: per-(b,hp) stats over 4096 elems + in_proj -> feat [4096][32]
// ---------------------------------------------------------------------------
__global__ __launch_bounds__(256) void stats_feat_kernel(
    const float* __restrict__ a, const float* __restrict__ b,
    const float* __restrict__ lemb, const float* __restrict__ ipw,
    const float* __restrict__ ipb, float* __restrict__ feat) {
  const int t   = blockIdx.x;          // (b,hp) pair
  const int tid = threadIdx.x;

  const float4* a4 = (const float4*)a + (size_t)t * (NPIX / 4);
  const float4* b4 = (const float4*)b + (size_t)t * (NPIX / 4);

  float sa = 0.f, sa2 = 0.f, sb = 0.f, sb2 = 0.f, sab = 0.f;
  float mna = 3.4e38f, mxa = -3.4e38f, mnb = 3.4e38f, mxb = -3.4e38f;

#pragma unroll
  for (int i = 0; i < 4; ++i) {
    float4 av = a4[tid + i * 256];
    float4 bv = b4[tid + i * 256];
    float ax[4] = {av.x, av.y, av.z, av.w};
    float bx[4] = {bv.x, bv.y, bv.z, bv.w};
#pragma unroll
    for (int c = 0; c < 4; ++c) {
      float x = ax[c], y = bx[c];
      sa += x; sa2 += x * x;
      sb += y; sb2 += y * y;
      sab += x * y;
      mna = fminf(mna, x); mxa = fmaxf(mxa, x);
      mnb = fminf(mnb, y); mxb = fmaxf(mxb, y);
    }
  }

  __shared__ float red[9][256];
  red[0][tid] = sa;  red[1][tid] = sa2; red[2][tid] = sb;  red[3][tid] = sb2;
  red[4][tid] = sab; red[5][tid] = mna; red[6][tid] = mxa; red[7][tid] = mnb;
  red[8][tid] = mxb;
  __syncthreads();

  for (int s = 128; s > 0; s >>= 1) {
    if (tid < s) {
      red[0][tid] += red[0][tid + s];
      red[1][tid] += red[1][tid + s];
      red[2][tid] += red[2][tid + s];
      red[3][tid] += red[3][tid + s];
      red[4][tid] += red[4][tid + s];
      red[5][tid] = fminf(red[5][tid], red[5][tid + s]);
      red[6][tid] = fmaxf(red[6][tid], red[6][tid + s]);
      red[7][tid] = fminf(red[7][tid], red[7][tid + s]);
      red[8][tid] = fmaxf(red[8][tid], red[8][tid + s]);
    }
    __syncthreads();
  }

  __shared__ float st[12];
  if (tid == 0) {
    const float N = (float)NPIX, NM1 = (float)(NPIX - 1);
    float Sa = red[0][0], Sa2 = red[1][0], Sb = red[2][0], Sb2 = red[3][0];
    float Sab = red[4][0];
    st[0] = Sa / N;
    st[1] = sqrtf(fmaxf((Sa2 - Sa * Sa / N) / NM1, 0.f));
    st[2] = red[5][0];
    st[3] = red[6][0];
    st[4] = Sb / N;
    st[5] = sqrtf(fmaxf((Sb2 - Sb * Sb / N) / NM1, 0.f));
    st[6] = red[7][0];
    st[7] = red[8][0];
    float Sd  = Sa - Sb;
    float Sd2 = fmaxf(Sa2 - 2.f * Sab + Sb2, 0.f);
    st[8]  = Sd / N;
    st[9]  = sqrtf(fmaxf((Sd2 - Sd * Sd / N) / NM1, 0.f));
    st[10] = sqrtf(Sd2);
    float na = fmaxf(sqrtf(Sa2), 1e-8f);
    float nb = fmaxf(sqrtf(Sb2), 1e-8f);
    st[11] = Sab / (na * nb);
  }
  __syncthreads();

  if (tid < D_) {
    float f = ipb[tid] + lemb[tid];
#pragma unroll
    for (int s = 0; s < 12; ++s) f += st[s] * ipw[tid * 12 + s];
    feat[t * D_ + tid] = f;
  }
}

// ---------------------------------------------------------------------------
// Kernel 2: transformer block per batch (64 tokens, D=32). 4 waves,
// one 16-token M-tile per wave; dense layers via v_wmma_f32_16x16x32_f16.
// ---------------------------------------------------------------------------
__global__ __launch_bounds__(128) void xf_kernel(
    const float* __restrict__ feat_in,
    const float* __restrict__ ln1_g, const float* __restrict__ ln1_b,
    const float* __restrict__ qkv_w, const float* __restrict__ qkv_b,
    const float* __restrict__ out_w, const float* __restrict__ out_b,
    const float* __restrict__ ln2_g, const float* __restrict__ ln2_b,
    const float* __restrict__ ffn_w1, const float* __restrict__ ffn_b1,
    const float* __restrict__ ffn_w2, const float* __restrict__ ffn_b2,
    const float* __restrict__ gate_w, const float* __restrict__ gate_b,
    const float* __restrict__ row_w, const float* __restrict__ row_b,
    const float* __restrict__ col_w, const float* __restrict__ col_b,
    const float* __restrict__ drow_w, const float* __restrict__ drow_b,
    const float* __restrict__ dcol_w, const float* __restrict__ dcol_b,
    float* __restrict__ gate_o, float* __restrict__ row_o,
    float* __restrict__ col_o, float* __restrict__ drow_o,
    float* __restrict__ dcol_o) {
  __shared__ float    s_feat[64][32];   // residual stream (f32)
  __shared__ float    s_buf[64][96];    // qkv (q|k|v)
  __shared__ _Float16 s_xh[64][32];     // f16 activations (LN1 / ao / LN2 / feat)
  __shared__ _Float16 s_hh[64][64];     // gelu(ffn1) f16

  const int bidx = blockIdx.x;          // batch
  const int tid  = threadIdx.x;
  const int lane = tid & 31;
  const int wave = tid >> 5;            // 0..3
  const int m0   = wave * 16;           // this wave's token tile

  // ---- load feat --------------------------------------------------------
  const float* fin = feat_in + (size_t)bidx * 64 * D_;
  for (int i = tid; i < 64 * D_; i += 128) (&s_feat[0][0])[i] = fin[i];
  __syncthreads();

  // ---- LN1 -> f16 --------------------------------------------------------
  if (tid < 64) {
    float mu = 0.f;
#pragma unroll
    for (int d = 0; d < D_; ++d) mu += s_feat[tid][d];
    mu *= (1.f / D_);
    float var = 0.f;
#pragma unroll
    for (int d = 0; d < D_; ++d) {
      float x = s_feat[tid][d] - mu;
      var += x * x;
    }
    float rs = rsqrtf(var * (1.f / D_) + 1e-5f);
#pragma unroll
    for (int d = 0; d < D_; ++d)
      s_xh[tid][d] = (_Float16)((s_feat[tid][d] - mu) * rs * ln1_g[d] + ln1_b[d]);
  }
  __syncthreads();

  // ---- qkv: [64,32] @ qkv_w[96,32]^T -> s_buf[64][96] --------------------
  {
    v16h af = frag_a_lds(&s_xh[0][0], 32, m0, 0, lane);
    for (int nt = 0; nt < 6; ++nt) {
      v16h bf = frag_b_wgt(qkv_w, 32, nt * 16, 0, lane);
      v8f c = {};
      c = wmma16(af, bf, c);
      int n  = nt * 16 + (lane & 15);
      int mb = m0 + ((lane >> 4) << 3);
      float bias = qkv_b[n];
#pragma unroll
      for (int r = 0; r < 8; ++r) s_buf[mb + r][n] = c[r] + bias;
    }
  }
  __syncthreads();

  // ---- attention (2 heads, DH=16) -> ao (f16) in s_xh --------------------
  {
    int h = tid >> 6, qi = tid & 63;
    float qv[16];
#pragma unroll
    for (int d = 0; d < 16; ++d) qv[d] = s_buf[qi][h * 16 + d];
    float mx = -3.4e38f;
    for (int kk = 0; kk < 64; ++kk) {
      float s = 0.f;
#pragma unroll
      for (int d = 0; d < 16; ++d) s += qv[d] * s_buf[kk][32 + h * 16 + d];
      mx = fmaxf(mx, s * 0.25f);
    }
    float Z = 0.f;
    float ao[16];
#pragma unroll
    for (int d = 0; d < 16; ++d) ao[d] = 0.f;
    for (int kk = 0; kk < 64; ++kk) {
      float s = 0.f;
#pragma unroll
      for (int d = 0; d < 16; ++d) s += qv[d] * s_buf[kk][32 + h * 16 + d];
      float p = __expf(s * 0.25f - mx);
      Z += p;
#pragma unroll
      for (int d = 0; d < 16; ++d) ao[d] += p * s_buf[kk][64 + h * 16 + d];
    }
    float iz = 1.f / Z;
#pragma unroll
    for (int d = 0; d < 16; ++d)
      s_xh[qi][h * 16 + d] = (_Float16)(ao[d] * iz);
  }
  __syncthreads();

  // ---- out proj: ao @ out_w[32,32]^T, residual add -----------------------
  {
    v16h af = frag_a_lds(&s_xh[0][0], 32, m0, 0, lane);
    for (int nt = 0; nt < 2; ++nt) {
      v16h bf = frag_b_wgt(out_w, 32, nt * 16, 0, lane);
      v8f c = {};
      c = wmma16(af, bf, c);
      int n  = nt * 16 + (lane & 15);
      int mb = m0 + ((lane >> 4) << 3);
      float bias = out_b[n];
#pragma unroll
      for (int r = 0; r < 8; ++r) s_feat[mb + r][n] += c[r] + bias;
    }
  }
  __syncthreads();

  // ---- LN2 -> f16 --------------------------------------------------------
  if (tid < 64) {
    float mu = 0.f;
#pragma unroll
    for (int d = 0; d < D_; ++d) mu += s_feat[tid][d];
    mu *= (1.f / D_);
    float var = 0.f;
#pragma unroll
    for (int d = 0; d < D_; ++d) {
      float x = s_feat[tid][d] - mu;
      var += x * x;
    }
    float rs = rsqrtf(var * (1.f / D_) + 1e-5f);
#pragma unroll
    for (int d = 0; d < D_; ++d)
      s_xh[tid][d] = (_Float16)((s_feat[tid][d] - mu) * rs * ln2_g[d] + ln2_b[d]);
  }
  __syncthreads();

  // ---- ffn1 + exact GELU -> s_hh (f16) -----------------------------------
  {
    v16h af = frag_a_lds(&s_xh[0][0], 32, m0, 0, lane);
    for (int nt = 0; nt < 4; ++nt) {
      v16h bf = frag_b_wgt(ffn_w1, 32, nt * 16, 0, lane);
      v8f c = {};
      c = wmma16(af, bf, c);
      int n  = nt * 16 + (lane & 15);
      int mb = m0 + ((lane >> 4) << 3);
      float bias = ffn_b1[n];
#pragma unroll
      for (int r = 0; r < 8; ++r) {
        float x = c[r] + bias;
        float g = 0.5f * x * (1.f + erff(x * 0.70710678118654752f));
        s_hh[mb + r][n] = (_Float16)g;
      }
    }
  }
  __syncthreads();

  // ---- ffn2 (K=64, two accumulating WMMAs), residual add -----------------
  {
    for (int nt = 0; nt < 2; ++nt) {
      v8f c = {};
#pragma unroll
      for (int kt = 0; kt < 2; ++kt) {
        v16h af = frag_a_lds(&s_hh[0][0], 64, m0, kt * 32, lane);
        v16h bf = frag_b_wgt(ffn_w2, 64, nt * 16, kt * 32, lane);
        c = wmma16(af, bf, c);
      }
      int n  = nt * 16 + (lane & 15);
      int mb = m0 + ((lane >> 4) << 3);
      float bias = ffn_b2[n];
#pragma unroll
      for (int r = 0; r < 8; ++r) s_feat[mb + r][n] += c[r] + bias;
    }
  }
  __syncthreads();

  // ---- final feat -> f16 -------------------------------------------------
  for (int i = tid; i < 64 * D_; i += 128)
    (&s_xh[0][0])[i] = (_Float16)((&s_feat[0][0])[i]);
  __syncthreads();

  // ---- output heads (row/col/drow/dcol), each [64,32]@[32,64]^T ----------
  {
    v16h af = frag_a_lds(&s_xh[0][0], 32, m0, 0, lane);
    const float* Ws[4] = {row_w, col_w, drow_w, dcol_w};
    const float* Bs[4] = {row_b, col_b, drow_b, dcol_b};
    float*       Os[4] = {row_o, col_o, drow_o, dcol_o};
#pragma unroll
    for (int w = 0; w < 4; ++w) {
      for (int nt = 0; nt < 4; ++nt) {
        v16h bf = frag_b_wgt(Ws[w], 32, nt * 16, 0, lane);
        v8f c = {};
        c = wmma16(af, bf, c);
        int n  = nt * 16 + (lane & 15);
        int mb = m0 + ((lane >> 4) << 3);
        float bias = Bs[w][n];
#pragma unroll
        for (int r = 0; r < 8; ++r)
          Os[w][((size_t)bidx * 64 + mb + r) * 64 + n] = c[r] + bias;
      }
    }
  }

  // ---- gate head (N=1, scalar) ------------------------------------------
  if (tid < 64) {
    float g = gate_b[0];
#pragma unroll
    for (int d = 0; d < D_; ++d) g += s_feat[tid][d] * gate_w[d];
    gate_o[(size_t)bidx * 64 + tid] = g;
  }
}

// ---------------------------------------------------------------------------
// Kernel 3: merged = sigmoid(gate+row+col)*a + (1-m)*b + ds*drow*dcol
// One block per (b,hp) tile; thread owns a 16-wide row segment.
// ---------------------------------------------------------------------------
__global__ __launch_bounds__(256) void merge_kernel(
    const float* __restrict__ a, const float* __restrict__ b,
    const float* __restrict__ gate_o, const float* __restrict__ row_o,
    const float* __restrict__ col_o, const float* __restrict__ drow_o,
    const float* __restrict__ dcol_o, const float* __restrict__ dscale,
    float* __restrict__ out) {
  const int t   = blockIdx.x;           // (b,hp)
  const int tid = threadIdx.x;
  const int i   = tid >> 2;             // row 0..63
  const int jb  = (tid & 3) << 4;       // col base (16-wide segment)

  const float g  = gate_o[t] + row_o[t * 64 + i];
  const float dr = dscale[0] * drow_o[t * 64 + i];

  const size_t base4 = (size_t)t * (NPIX / 4) + (size_t)i * 16 + (jb >> 2);
  const float4* a4 = (const float4*)a + base4;
  const float4* b4 = (const float4*)b + base4;
  float4*       o4 = (float4*)out + base4;

#pragma unroll
  for (int q = 0; q < 4; ++q) {
    float4 av = a4[q];
    float4 bv = b4[q];
    float ax[4] = {av.x, av.y, av.z, av.w};
    float bx[4] = {bv.x, bv.y, bv.z, bv.w};
    float ox[4];
#pragma unroll
    for (int c = 0; c < 4; ++c) {
      int jj = jb + q * 4 + c;
      float ml   = g + col_o[t * 64 + jj];
      float mask = 1.f / (1.f + __expf(-ml));
      float dlt  = dr * dcol_o[t * 64 + jj];
      ox[c] = mask * ax[c] + (1.f - mask) * bx[c] + dlt;
    }
    o4[q] = make_float4(ox[0], ox[1], ox[2], ox[3]);
  }
}

// ---------------------------------------------------------------------------
// Launch
// ---------------------------------------------------------------------------
extern "C" void kernel_launch(void* const* d_in, const int* in_sizes, int n_in,
                              void* d_out, int out_size, void* d_ws,
                              size_t ws_size, hipStream_t stream) {
  const float* a        = (const float*)d_in[0];
  const float* b        = (const float*)d_in[1];
  const float* lemb     = (const float*)d_in[2];
  const float* ipw      = (const float*)d_in[3];
  const float* ipb      = (const float*)d_in[4];
  const float* ln1_g    = (const float*)d_in[5];
  const float* ln1_b    = (const float*)d_in[6];
  const float* qkv_w    = (const float*)d_in[7];
  const float* qkv_b    = (const float*)d_in[8];
  const float* out_w    = (const float*)d_in[9];
  const float* out_b    = (const float*)d_in[10];
  const float* ln2_g    = (const float*)d_in[11];
  const float* ln2_b    = (const float*)d_in[12];
  const float* ffn_w1   = (const float*)d_in[13];
  const float* ffn_b1   = (const float*)d_in[14];
  const float* ffn_w2   = (const float*)d_in[15];
  const float* ffn_b2   = (const float*)d_in[16];
  const float* gate_w   = (const float*)d_in[17];
  const float* gate_b   = (const float*)d_in[18];
  const float* row_w    = (const float*)d_in[19];
  const float* row_b    = (const float*)d_in[20];
  const float* col_w    = (const float*)d_in[21];
  const float* col_b    = (const float*)d_in[22];
  const float* drow_w   = (const float*)d_in[23];
  const float* drow_b   = (const float*)d_in[24];
  const float* dcol_w   = (const float*)d_in[25];
  const float* dcol_b   = (const float*)d_in[26];
  const float* dscale   = (const float*)d_in[27];

  float* ws     = (float*)d_ws;
  float* feat   = ws;                       // 4096*32   = 131072
  float* gate_o = feat + NTOK * D_;         // 4096
  float* row_o  = gate_o + NTOK;            // 4096*64   = 262144
  float* col_o  = row_o + NTOK * H_;
  float* drow_o = col_o + NTOK * W_;
  float* dcol_o = drow_o + NTOK * H_;       // total ~4.6 MB

  float* out = (float*)d_out;

  stats_feat_kernel<<<NTOK, 256, 0, stream>>>(a, b, lemb, ipw, ipb, feat);

  xf_kernel<<<B_, 128, 0, stream>>>(
      feat, ln1_g, ln1_b, qkv_w, qkv_b, out_w, out_b, ln2_g, ln2_b, ffn_w1,
      ffn_b1, ffn_w2, ffn_b2, gate_w, gate_b, row_w, row_b, col_w, col_b,
      drow_w, drow_b, dcol_w, dcol_b, gate_o, row_o, col_o, drow_o, dcol_o);

  merge_kernel<<<NTOK, 256, 0, stream>>>(a, b, gate_o, row_o, col_o, drow_o,
                                         dcol_o, dscale, out);
}
